// EnsembleDynamicsGradAttn_44006234914943
// MI455X (gfx1250) — matrix operand order repaired and checked by reading, not compiled, not run on hardware
//
#include <hip/hip_runtime.h>
#include <hip/hip_bf16.h>

typedef __attribute__((ext_vector_type(16))) __bf16 v16bf;
typedef __attribute__((ext_vector_type(8)))  __bf16 v8bf;
typedef __attribute__((ext_vector_type(8)))  float  v8f;

#define E_ 5
#define C_ 32
#define N_ 16
#define B_ 256
#define IN_ 32
#define F_ 64
#define H_ 4
#define AK_ 32

// pack two f32 into bf16x2 (truncation) with a single v_perm_b32
__device__ __forceinline__ unsigned pk2(float a, float b) {
  union { float f; unsigned u; } x, y; x.f = a; y.f = b;
  // D = { b[31:16], a[31:16] } : sel bytes {7,6,3,2} of {S0=b, S1=a}
  return __builtin_amdgcn_perm(y.u, x.u, 0x07060302u);
}
__device__ __forceinline__ unsigned short f2bf_hi(float f) {
  union { float f; unsigned u; } v; v.f = f;
  return (unsigned short)(v.u >> 16);  // folds into ds_store_b16_d16_hi
}
__device__ __forceinline__ float bf2f(unsigned short s) {
  union { unsigned u; float f; } v; v.u = ((unsigned)s) << 16; return v.f;
}
// 16 contiguous f32 -> v16bf (identity order)
__device__ __forceinline__ v16bf cvt16(const float* p) {
  union { v16bf v; unsigned u[8]; } r;
#pragma unroll
  for (int i = 0; i < 8; ++i) r.u[i] = pk2(p[2 * i], p[2 * i + 1]);
  return r.v;
}

// ---- A fragment (16x32 MxK bf16): lane holds row M=L%16,
// a[i] <-> K = (L/16)*16 + i (matches B fragment order).
// K-runs contiguous: a[0..15] = K half*16..+15.
__device__ __forceinline__ v16bf load_A_lds(const unsigned short* p, int st,
                                            int c0, int lane) {
  int row = lane & 15, half = lane >> 4;
  const __bf16* base = (const __bf16*)(p) + row * st + c0 + half * 16;
  union { v16bf v; v8bf h[2]; } a;
  a.h[0] = *(const v8bf*)(base);       // K = half*16 + 0..7
  a.h[1] = *(const v8bf*)(base + 8);   // K = half*16 + 8..15
  return a.v;
}
// from f32 source via accessor f(row,k), k contiguous (loads merge to b128)
template <typename FN>
__device__ __forceinline__ v16bf load_A_g32(FN f, int lane) {
  int row = lane & 15, half = lane >> 4;
  union { v16bf v; unsigned u[8]; } a;
#pragma unroll
  for (int j = 0; j < 8; ++j) {
    int k = (half << 4) + (j << 1);
    a.u[j] = pk2(f(row, k), f(row, k + 1));
  }
  return a.v;
}

// ---- B fragment (32x16 KxN bf16): lane holds col N=L%16,
// b[i] <-> K = (L/16)*16 + i, i.e. 16 contiguous K per lane.
__device__ __forceinline__ v16bf load_B_tr(const unsigned short* pt, int st,
                                           int n0, int k0, int lane) {
  int col = lane & 15, half = lane >> 4;
  const __bf16* base = (const __bf16*)(pt) + (n0 + col) * st + k0 + half * 16;
  union { v16bf v; v8bf h[2]; } b;
  b.h[0] = *(const v8bf*)(base);
  b.h[1] = *(const v8bf*)(base + 8);
  return b.v;
}
// from f32 K-major source via accessor f(k,col) (coalesced across lanes)
template <typename FN>
__device__ __forceinline__ v16bf load_B_g32(FN f, int lane) {
  int col = lane & 15, half = lane >> 4;
  union { v16bf v; unsigned u[8]; } b;
#pragma unroll
  for (int j = 0; j < 8; ++j)
    b.u[j] = pk2(f((half << 4) + 2 * j, col), f((half << 4) + 2 * j + 1, col));
  return b.v;
}

__device__ __forceinline__ v8f wmma_bf16(v16bf a, v16bf b, v8f c) {
  return __builtin_amdgcn_wmma_f32_16x16x32_bf16(false, a, false, b, (short)0, c,
                                                 false, false);
}

// D store to row-major bf16 LDS: dst[(r+8*half)*st + col]; zero-VALU stores
__device__ __forceinline__ void store_D(unsigned short* dst, int st, int col,
                                        int half, v8f acc, float bias,
                                        bool relu) {
#pragma unroll
  for (int r = 0; r < 8; ++r) {
    float v = acc[r] + bias;
    if (relu) v = fmaxf(v, 0.f);
    dst[(r + 8 * half) * st + col] = f2bf_hi(v);
  }
}
// D store to N-major LDS: dst[col*st + row0 + r + 8*half]; packed b32 stores
__device__ __forceinline__ void store_D_tr(unsigned short* dst, int st, int row0,
                                           int col, int half, v8f acc) {
  unsigned* base = (unsigned*)(dst + col * st + row0 + 8 * half);
#pragma unroll
  for (int r = 0; r < 8; r += 2) base[r >> 1] = pk2(acc[r], acc[r + 1]);
}

__global__ __launch_bounds__(64) void fused_ensemble_kernel(
    const float* __restrict__ x,
    const float* __restrict__ fw0, const float* __restrict__ fb0,
    const float* __restrict__ fw1, const float* __restrict__ fb1,
    const float* __restrict__ wq0, const float* __restrict__ wk0,
    const float* __restrict__ wv0, const float* __restrict__ wo0,
    const float* __restrict__ wq1, const float* __restrict__ wk1,
    const float* __restrict__ wv1, const float* __restrict__ wo1,
    const float* __restrict__ pw0, const float* __restrict__ pb0,
    const float* __restrict__ pw1, const float* __restrict__ pb1,
    const int* __restrict__ qidx,
    float* __restrict__ out) {
  // block-shared folded attention weights, stored N-MAJOR (transposed)
  __shared__ __align__(16) unsigned short WqkT[F_ * F_];  // 8 KB
  __shared__ __align__(16) unsigned short WvoT[F_ * F_];  // 8 KB
  // per-wave ping-pong h buffers [n][b_local][d] bf16
  __shared__ __align__(16) unsigned short hbuf[2][2][N_ * 16 * F_];  // 128 KB
  __shared__ __align__(16) unsigned short tub[2][16 * F_];  // T scratch, 4 KB
  __shared__ __align__(16) unsigned short utb[2][64 * 32];  // U (64x32), 8 KB
  __shared__ __align__(16) unsigned short att[2][16 * 32];  // attn 16x32, 2 KB

  const int lane = threadIdx.x & 31;
  const int wave = threadIdx.x >> 5;
  const int bt = blockIdx.x & 7;
  const int c  = (blockIdx.x >> 3) & 31;
  const int e  = blockIdx.x >> 8;
  const int ec = e * C_ + c;
  const int b0 = bt * 32 + wave * 16;
  const int half = lane >> 4;
  const int lcol = lane & 15;

  unsigned short* hA = hbuf[wave][0];
  unsigned short* hB = hbuf[wave][1];
  unsigned short* TU = tub[wave];  // 16x64 row-major
  unsigned short* UT = utb[wave];  // 64x32 N-major, K in [16,32) stays zero
  unsigned short* AT = att[wave];  // 16x32 row-major, K in [16,32) stays zero

  // one-time zero of the K-padding regions (per-wave buffers, no barrier)
  {
    const uint4 z = {0u, 0u, 0u, 0u};
    for (int o = lane; o < 64; o += 32) {
      *(uint4*)(UT + o * 32 + 16) = z;
      *(uint4*)(UT + o * 32 + 24) = z;
    }
    if (lane < 16) {
      *(uint4*)(AT + lane * 32 + 16) = z;
      *(uint4*)(AT + lane * 32 + 24) = z;
    }
  }

  // ------------------ Layer 1 + Layer 2 (per node) ------------------
  const float* fw0_ec = fw0 + (size_t)ec * N_ * IN_ * F_;
  const float* fb0_ec = fb0 + (size_t)ec * N_ * F_;
  const float* fw1_ec = fw1 + (size_t)ec * N_ * F_ * F_;
  const float* fb1_ec = fb1 + (size_t)ec * N_ * F_;

  for (int n = 0; n < N_; ++n) {
    const float* w0n = fw0_ec + (size_t)n * IN_ * F_;
    const float* b0n = fb0_ec + (size_t)n * F_;
    const float* w1n = fw1_ec + (size_t)n * F_ * F_;
    const float* b1n = fb1_ec + (size_t)n * F_;
    if (n + 1 < N_) {  // prefetch next node's weights into L2
      __builtin_prefetch(w0n + IN_ * F_, 0, 0);
      __builtin_prefetch(w1n + F_ * F_, 0, 0);
    }
    // layer1: [16b x 32in] @ [32in x 64f]
    v16bf a1 = load_A_g32(
        [&](int r, int k) -> float {
          return x[((size_t)n * B_ + (b0 + r)) * IN_ + k];
        },
        lane);
    for (int nt = 0; nt < 4; ++nt) {
      v16bf bw = load_B_g32(
          [&](int k, int cc) -> float {
            return w0n[(size_t)k * F_ + nt * 16 + cc];
          },
          lane);
      v8f acc = {};
      acc = wmma_bf16(a1, bw, acc);
      int col = nt * 16 + lcol;
      store_D(TU, F_, col, half, acc, b0n[col], true);
    }
    // layer2: [16b x 64] @ [64 x 64]
    v16bf a2[2];
#pragma unroll
    for (int kc = 0; kc < 2; ++kc) a2[kc] = load_A_lds(TU, F_, kc * 32, lane);
    for (int nt = 0; nt < 4; ++nt) {
      v8f acc = {};
#pragma unroll
      for (int kc = 0; kc < 2; ++kc) {
        v16bf bw = load_B_g32(
            [&](int k, int cc) -> float {
              return w1n[(size_t)(kc * 32 + k) * F_ + nt * 16 + cc];
            },
            lane);
        acc = wmma_bf16(a2[kc], bw, acc);
      }
      int col = nt * 16 + lcol;
      store_D(hA + (size_t)n * 16 * F_, F_, col, half, acc, b1n[col], true);
    }
  }

  // ------------------ Two MHA blocks over the node axis ------------------
  for (int st = 0; st < 2; ++st) {
    const float* wqp = (st ? wq1 : wq0) + (size_t)ec * H_ * F_ * AK_;
    const float* wkp = (st ? wk1 : wk0) + (size_t)ec * H_ * F_ * AK_;
    const float* wvp = (st ? wv1 : wv0) + (size_t)ec * H_ * F_ * AK_;
    const float* wop = (st ? wo1 : wo0) + (size_t)ec * (H_ * AK_) * F_;

    // residual seed: hB = hA (16B vector copies)
    {
      const uint4* s = (const uint4*)hA;
      uint4* d = (uint4*)hB;
      for (int i = lane; i < N_ * 16 * F_ / 8; i += 32) d[i] = s[i];
    }

    for (int h = 0; h < H_; ++h) {
      __syncthreads();  // previous head/stage done reading WqkT/WvoT
      {
        // wave0: WqkT = (Wq @ Wk^T)^T ; wave1: WvoT = (Wv @ Wo_h)^T
        const float* wa = (wave ? wvp : wqp) + (size_t)h * F_ * AK_;
        unsigned short* Wdst = wave ? WvoT : WqkT;
        for (int mt = 0; mt < 4; ++mt) {
          v16bf a = load_A_g32(
              [&](int r, int k) -> float {
                return wa[(size_t)(mt * 16 + r) * AK_ + k];
              },
              lane);
          for (int nt = 0; nt < 4; ++nt) {
            v16bf b;
            if (wave == 0) {
              // Wk^T[k][d'] = wk[d'][k]: K-contiguous f32 run per lane
              const float* p =
                  wkp + (size_t)(h * F_ + nt * 16 + lcol) * AK_ + half * 16;
              b = cvt16(p);
            } else {
              b = load_B_g32(
                  [&](int k, int cc) -> float {  // wo rows h*32..h*32+31
                    return wop[(size_t)(h * AK_ + k) * F_ + nt * 16 + cc];
                  },
                  lane);
            }
            v8f acc = {};
            acc = wmma_bf16(a, b, acc);
            store_D_tr(Wdst, F_, mt * 16, nt * 16 + lcol, half, acc);
          }
        }
      }
      __syncthreads();

      for (int b = 0; b < 16; ++b) {
        // H_b row fragments (reused by T and U GEMMs)
        v16bf ha[2];
#pragma unroll
        for (int kc = 0; kc < 2; ++kc)
          ha[kc] = load_A_lds(hA + b * F_, 16 * F_, kc * 32, lane);
        // T_b = H_b @ Wqk -> TU row-major (16 x 64)
        for (int nt = 0; nt < 4; ++nt) {
          v8f acc = {};
#pragma unroll
          for (int kc = 0; kc < 2; ++kc)
            acc = wmma_bf16(ha[kc],
                            load_B_tr(WqkT, F_, nt * 16, kc * 32, lane), acc);
          store_D(TU, F_, nt * 16 + lcol, half, acc, 0.f, false);
        }
        // scores_b = T_b @ H_b^T (16q x 16m); H_b^T is K-fast in hA
        v8f sc = {};
#pragma unroll
        for (int kc = 0; kc < 2; ++kc) {
          v16bf at = load_A_lds(TU, F_, kc * 32, lane);
          v16bf bh = load_B_tr(hA + b * F_, 16 * F_, 0, kc * 32, lane);
          sc = wmma_bf16(at, bh, sc);
        }
        // softmax over m (one column per lane within each 16-lane half)
        const float scl = 0.17677669529663687f;  // 1/sqrt(32)
#pragma unroll
        for (int r = 0; r < 8; ++r) {
          float v = sc[r] * scl;
          float mx = v;
#pragma unroll
          for (int m = 1; m < 16; m <<= 1) mx = fmaxf(mx, __shfl_xor(mx, m, 32));
          float ex = __expf(v - mx);
          float sm = ex;
#pragma unroll
          for (int m = 1; m < 16; m <<= 1) sm += __shfl_xor(sm, m, 32);
          AT[(r + 8 * half) * 32 + lcol] = f2bf_hi(ex / sm);
        }
        // U_b = H_b @ Wvo -> UT N-major (64 x 32, K>=16 pre-zeroed)
        for (int nt = 0; nt < 4; ++nt) {
          v8f acc = {};
#pragma unroll
          for (int kc = 0; kc < 2; ++kc)
            acc = wmma_bf16(ha[kc],
                            load_B_tr(WvoT, F_, nt * 16, kc * 32, lane), acc);
          store_D_tr(UT, 32, 0, nt * 16 + lcol, half, acc);
        }
        // hB[n][b][:] += attn_b @ U_b  (K=16, padding pre-zeroed in LDS)
        v16bf aat = load_A_lds(AT, 32, 0, lane);
        for (int nt = 0; nt < 4; ++nt) {
          int col = nt * 16 + lcol;
          v16bf bu = load_B_tr(UT, 32, nt * 16, 0, lane);
          v8f cacc;
#pragma unroll
          for (int r = 0; r < 8; ++r)
            cacc[r] = bf2f(hB[((r + 8 * half) * 16 + b) * F_ + col]);
          cacc = wmma_bf16(aat, bu, cacc);
          store_D(hB + b * F_, 16 * F_, col, half, cacc, 0.f, false);
        }
      }  // b
    }    // heads
    unsigned short* t = hA; hA = hB; hB = t;  // ping-pong
  }  // stages

  // ------------------ Gather query node + 2-layer head ------------------
  int qn = qidx[c];
  qn = (qn < 0) ? 0 : (qn >= N_ ? N_ - 1 : qn);
  const float* pw0ec = pw0 + (size_t)ec * F_ * F_;
  const float* pb0ec = pb0 + (size_t)ec * F_;
  const float* pw1ec = pw1 + (size_t)ec * F_;
  const float* pb1ec = pb1 + (size_t)ec;

  v16bf aq[2];
#pragma unroll
  for (int kc = 0; kc < 2; ++kc)
    aq[kc] = load_A_lds(hA + (size_t)qn * 16 * F_, F_, kc * 32, lane);
  for (int nt = 0; nt < 4; ++nt) {
    v8f acc = {};
#pragma unroll
    for (int kc = 0; kc < 2; ++kc) {
      v16bf bw = load_B_g32(
          [&](int k, int cc) -> float {
            return pw0ec[(size_t)(kc * 32 + k) * F_ + nt * 16 + cc];
          },
          lane);
      acc = wmma_bf16(aq[kc], bw, acc);
    }
    int col = nt * 16 + lcol;
    store_D(TU, F_, col, half, acc, pb0ec[col], true);
  }
  // final 64 -> 1 projection: one batch row per lane (lanes 0..15)
  if (lane < 16) {
    float s = pb1ec[0];
    for (int d = 0; d < F_; ++d) s += bf2f(TU[lane * F_ + d]) * pw1ec[d];
    out[(size_t)ec * B_ + b0 + lane] = s;
  }
}

extern "C" void kernel_launch(void* const* d_in, const int* in_sizes, int n_in,
                              void* d_out, int out_size, void* d_ws,
                              size_t ws_size, hipStream_t stream) {
  const float* x   = (const float*)d_in[0];
  const float* fw0 = (const float*)d_in[1];
  const float* fb0 = (const float*)d_in[2];
  const float* fw1 = (const float*)d_in[3];
  const float* fb1 = (const float*)d_in[4];
  const float* wq0 = (const float*)d_in[5];
  const float* wk0 = (const float*)d_in[6];
  const float* wv0 = (const float*)d_in[7];
  const float* wo0 = (const float*)d_in[8];
  const float* wq1 = (const float*)d_in[9];
  const float* wk1 = (const float*)d_in[10];
  const float* wv1 = (const float*)d_in[11];
  const float* wo1 = (const float*)d_in[12];
  const float* pw0 = (const float*)d_in[13];
  const float* pb0 = (const float*)d_in[14];
  const float* pw1 = (const float*)d_in[15];
  const float* pb1 = (const float*)d_in[16];
  const int*  qidx = (const int*)d_in[17];
  float* out = (float*)d_out;

  dim3 grid(E_ * C_ * (B_ / 32));  // 1280 blocks: (e, c, batch tile of 32)
  dim3 block(64);                  // 2 waves; each wave owns 16 batch rows
  fused_ensemble_kernel<<<grid, block, 0, stream>>>(
      x, fw0, fb0, fw1, fb1, wq0, wk0, wv0, wo0, wq1, wk1, wv1, wo1, pw0, pb0,
      pw1, pb1, qidx, out);
}
